// PredictiveModule_31421980737633
// MI455X (gfx1250) — compile-verified, matrix operand order checked by reading, not run on hardware
//
#include <hip/hip_runtime.h>
#include <hip/hip_bf16.h>
#include <math.h>

#define THREADS 256

// ---- compile-time WMMA path selection (device codegen pass only) ----------
#if defined(__HIP_DEVICE_COMPILE__)
# if __has_builtin(__builtin_amdgcn_wmma_f32_16x16x4_f32)
#  define PM_WMMA_F32X4 1
# elif __has_builtin(__builtin_amdgcn_wmma_f32_16x16x32_f16)
#  define PM_WMMA_F16X32 1
# endif
#endif

typedef __attribute__((ext_vector_type(2)))  float    pm_v2f;
typedef __attribute__((ext_vector_type(8)))  float    pm_v8f;
typedef __attribute__((ext_vector_type(16))) _Float16 pm_v16h;

// ---------------------------------------------------------------------------
// init: accumulators <- bias, logprob accumulator <- 0
// ---------------------------------------------------------------------------
__global__ void pm_init_kernel(const float* __restrict__ b1,
                               const float* __restrict__ b2,
                               const float* __restrict__ b3,
                               float* __restrict__ acc1,
                               float* __restrict__ acc2,
                               float* __restrict__ acc3,
                               float* __restrict__ lp,
                               int H, int D) {
    int i = blockIdx.x * blockDim.x + threadIdx.x;
    if (i < H) {
        acc1[i] = b1[i];
        acc2[i] = b2[i];
    }
    if (i < D) {
        acc3[i] = b3[i];
    }
    if (i == 0) *lp = 0.f;
}

// ---------------------------------------------------------------------------
// Split-K GEMV: acc[N] += x[K-slice] @ W[K,N] (row-major W).
// One wave = one 16-column tile x one K-slice of kpw rows; partials are
// reduced with global_atomic_add_f32. The x vector rides in row M=0 of the
// WMMA A operand; only C row 0 (VGPR0, lanes 0-15) is read back.
// ---------------------------------------------------------------------------
__global__ void pm_gemv_wmma_kernel(const float* __restrict__ W,
                                    const float* __restrict__ x,
                                    float* __restrict__ acc,
                                    int N, int coltiles, int kpw,
                                    int total_waves) {
    const int gtid = blockIdx.x * blockDim.x + threadIdx.x;
    const int w    = gtid >> 5;
    const int lane = threadIdx.x & 31;
    if (w >= total_waves) return;

    const int col_tile = w % coltiles;     // adjacent waves -> adjacent tiles
    const int kslice   = w / coltiles;
    const int k0       = kslice * kpw;
    const int col_base = col_tile << 4;

    const int  n    = lane & 15;           // column within tile
    const bool is_a = (n == 0);            // lanes 0 / 16 carry the x vector

    const float* __restrict__ wcol = W + (size_t)col_base + (size_t)n;

    float result;                          // partial y[col_base + n], lanes 0-15

#if defined(PM_WMMA_F32X4)
    // ---- f32 path: V_WMMA_F32_16X16X4_F32, K step 4, unrolled x8 ---------
    // A 16x4 f32: v0 lanes0-15 K=0 / lanes16-31 K=2; v1 K=1 / K=3.
    // B 4x16 f32: v0 lanes0-15 row K=0 / lanes16-31 row K=2; v1 K=1 / K=3.
    const int khalf = (lane >> 4) << 1;    // 0 or 2
    pm_v8f c = {};
    for (int kb = k0; kb < k0 + kpw; kb += 32) {
        // pull the weight stream ~64 rows ahead (global_prefetch_b8)
        __builtin_prefetch(wcol + (size_t)(kb + 64) * (size_t)N, 0, 1);
#pragma unroll
        for (int u = 0; u < 8; ++u) {
            const int kr = kb + 4 * u + khalf;
            const float* wp = wcol + (size_t)kr * (size_t)N;
            pm_v2f b;
            b.x = wp[0];
            b.y = wp[N];
            const float x0 = x[kr];
            const float x1 = x[kr + 1];
            pm_v2f a;
            a.x = is_a ? x0 : 0.f;
            a.y = is_a ? x1 : 0.f;
            c = __builtin_amdgcn_wmma_f32_16x16x4_f32(
                    false, a, false, b, (short)0, c, false, false);
        }
    }
    result = c[0];

#elif defined(PM_WMMA_F16X32)
    // ---- f16 path: V_WMMA_F32_16X16X32_F16, K step 32 --------------------
    // 16-bit A 16x32: lane group 0 (lanes 0-15) holds K=0..7,16..23;
    // lane group 1 (lanes 16-31) holds K=8..15,24..31 (half h -> K offset
    // ko = (h&7) + 16*(h>>3) + 8*group). B 32x16 mirrors the same K map per
    // column lane. Weights converted fp32->fp16 in-register (HBM bytes equal).
    const int grp8 = (lane >> 4) << 3;     // 0 or 8
    pm_v8f c = {};
#pragma unroll 2
    for (int k = k0; k < k0 + kpw; k += 32) {
        if (((k - k0) & 63) == 0)
            __builtin_prefetch(wcol + (size_t)(k + 64) * (size_t)N, 0, 1);
        pm_v16h ah, bh;
#pragma unroll
        for (int h = 0; h < 16; ++h) {
            const int ko = (h & 7) + ((h >> 3) << 4) + grp8;
            const float wv = wcol[(size_t)(k + ko) * (size_t)N];
            bh[h] = (_Float16)wv;
            const float xv = x[k + ko];
            ah[h] = is_a ? (_Float16)xv : (_Float16)0.f;
        }
        c = __builtin_amdgcn_wmma_f32_16x16x32_f16(
                false, ah, false, bh, (short)0, c, false, false);
    }
    result = c[0];

#else
    // ---- VALU fallback (also covers the host semantic pass) --------------
    const int g = lane >> 4;
    float sum = 0.f;
    for (int k = k0 + g; k < k0 + kpw; k += 2) {
        sum = fmaf(x[k], wcol[(size_t)k * (size_t)N], sum);
    }
    sum += __shfl_xor(sum, 16, 32);
    result = sum;
#endif

    if (lane < 16) {
        atomicAdd(&acc[col_base + n], result);   // global_atomic_add_f32
    }
}

// ---------------------------------------------------------------------------
// relu: h = max(acc, 0)
// ---------------------------------------------------------------------------
__global__ void pm_relu_kernel(const float* __restrict__ acc,
                               float* __restrict__ h, int n) {
    int i = blockIdx.x * blockDim.x + threadIdx.x;
    if (i < n) h[i] = fmaxf(acc[i], 0.f);
}

// ---------------------------------------------------------------------------
// elementwise tail: clip/gate/sample + logprob partial reduction
// ---------------------------------------------------------------------------
__global__ void pm_sample_kernel(const float* __restrict__ acc3,
                                 const float* __restrict__ ct,
                                 const float* __restrict__ qm,
                                 const float* __restrict__ noise,
                                 float* __restrict__ sample,
                                 float* __restrict__ absv,
                                 float* __restrict__ lp) {
    __shared__ float red[THREADS];
    const int i = blockIdx.x * blockDim.x + threadIdx.x;

    float vm = acc3[i];
    vm = fminf(fmaxf(vm, -100.f), 100.f);                     // output clip
    const float g = 1.f / (1.f + __expf(-(qm[i] * ct[i])));   // sigmoid gate
    // influence = clip(1 + tanh(0), 0.1, 10) == 1.0 exactly
    float am = g * vm;
    am = fminf(fmaxf(am, -1000.f), 1000.f);                   // mean clip
    const float s = am + 0.05f * noise[i];
    sample[i] = s;
    absv[i]   = fabsf(s);

    const float z = (s - am) * 20.f;                          // / ACTION_STD
    // -log(0.05) - 0.5*log(2*pi) = 2.0767937...
    const float l = -0.5f * z * z + 2.0767937f;

    red[threadIdx.x] = l;
    __syncthreads();
    for (int off = THREADS / 2; off > 0; off >>= 1) {
        if (threadIdx.x < off) red[threadIdx.x] += red[threadIdx.x + off];
        __syncthreads();
    }
    if (threadIdx.x == 0) atomicAdd(lp, red[0]);
}

// ---------------------------------------------------------------------------
// top-k (k of D) by |sample|: bisection threshold + exact-k tie fill.
// Single 1024-thread block. Also emits log_prob to out[D].
// ---------------------------------------------------------------------------
__global__ void pm_topk_kernel(const float* __restrict__ sample,
                               const float* __restrict__ absv,
                               const float* __restrict__ lp,
                               float* __restrict__ out,
                               int D, const int* __restrict__ kp) {
    __shared__ float red[1024];
    __shared__ float s_lo, s_hi;
    __shared__ int   s_cnt, s_ticket;

    const int t = threadIdx.x;
    const int k = *kp;

    float m = 0.f;
    for (int i = t; i < D; i += 1024) m = fmaxf(m, absv[i]);
    red[t] = m;
    __syncthreads();
    for (int off = 512; off > 0; off >>= 1) {
        if (t < off) red[t] = fmaxf(red[t], red[t + off]);
        __syncthreads();
    }
    if (t == 0) { s_lo = 0.f; s_hi = red[0]; }
    __syncthreads();

    // invariant: count(>lo) >= k, count(>hi) < k
    for (int it = 0; it < 40; ++it) {
        const float mid = 0.5f * (s_lo + s_hi);
        int c = 0;
        for (int i = t; i < D; i += 1024) c += (absv[i] > mid) ? 1 : 0;
        red[t] = (float)c;
        __syncthreads();
        for (int off = 512; off > 0; off >>= 1) {
            if (t < off) red[t] += red[t + off];
            __syncthreads();
        }
        if (t == 0) {
            if ((int)red[0] >= k) s_lo = mid; else s_hi = mid;
        }
        __syncthreads();
    }

    const float thr = s_hi;
    int c = 0;
    for (int i = t; i < D; i += 1024) c += (absv[i] > thr) ? 1 : 0;
    red[t] = (float)c;
    __syncthreads();
    for (int off = 512; off > 0; off >>= 1) {
        if (t < off) red[t] += red[t + off];
        __syncthreads();
    }
    if (t == 0) { s_cnt = (int)red[0]; s_ticket = 0; }
    __syncthreads();

    const int   extra = k - s_cnt;     // ties to take from (lo, thr]
    const float lo    = s_lo;
    for (int i = t; i < D; i += 1024) {
        const float a = absv[i];
        float o = 0.f;
        if (a > thr) {
            o = sample[i];
        } else if (a > lo && extra > 0) {
            const int tk = atomicAdd(&s_ticket, 1);
            if (tk < extra) o = sample[i];
        }
        out[i] = o;
    }
    if (t == 0) out[D] = *lp;          // log_prob
}

// ---------------------------------------------------------------------------
extern "C" void kernel_launch(void* const* d_in, const int* in_sizes, int n_in,
                              void* d_out, int out_size, void* d_ws, size_t ws_size,
                              hipStream_t stream) {
    const float* ct    = (const float*)d_in[0];
    const float* W1    = (const float*)d_in[1];
    const float* b1    = (const float*)d_in[2];
    const float* W2    = (const float*)d_in[3];
    const float* b2    = (const float*)d_in[4];
    const float* W3    = (const float*)d_in[5];
    const float* b3    = (const float*)d_in[6];
    const float* qm    = (const float*)d_in[7];
    const float* noise = (const float*)d_in[8];
    const int*   kptr  = (const int*)d_in[9];

    const int D = in_sizes[0];   // 16384
    const int H = in_sizes[2];   // 4096

    float* ws     = (float*)d_ws;
    float* acc1   = ws;              // [H]
    float* h1     = acc1 + H;        // [H]
    float* acc2   = h1 + H;          // [H]
    float* h2     = acc2 + H;        // [H]
    float* acc3   = h2 + H;          // [D]
    float* sample = acc3 + D;        // [D]
    float* absv   = sample + D;      // [D]
    float* lp     = absv + D;        // [1]

    float* out = (float*)d_out;      // [D] write_vec + [1] log_prob

    pm_init_kernel<<<(D + THREADS - 1) / THREADS, THREADS, 0, stream>>>(
        b1, b2, b3, acc1, acc2, acc3, lp, H, D);

    const int kpw = 512;             // K rows per wave (split-K slice)

    // layer 1: acc1 += ct @ W1   (K=D, N=H) -> 256 tiles x 32 slices
    {
        const int N = H, K = D;
        const int coltiles = N / 16, splitk = K / kpw;
        const int waves = coltiles * splitk;
        pm_gemv_wmma_kernel<<<(waves * 32 + THREADS - 1) / THREADS, THREADS, 0, stream>>>(
            W1, ct, acc1, N, coltiles, kpw, waves);
        pm_relu_kernel<<<(N + THREADS - 1) / THREADS, THREADS, 0, stream>>>(acc1, h1, N);
    }
    // layer 2: acc2 += h1 @ W2   (K=H, N=H)
    {
        const int N = H, K = H;
        const int coltiles = N / 16, splitk = K / kpw;
        const int waves = coltiles * splitk;
        pm_gemv_wmma_kernel<<<(waves * 32 + THREADS - 1) / THREADS, THREADS, 0, stream>>>(
            W2, h1, acc2, N, coltiles, kpw, waves);
        pm_relu_kernel<<<(N + THREADS - 1) / THREADS, THREADS, 0, stream>>>(acc2, h2, N);
    }
    // layer 3: acc3 += h2 @ W3   (K=H, N=D)
    {
        const int N = D, K = H;
        const int coltiles = N / 16, splitk = K / kpw;
        const int waves = coltiles * splitk;
        pm_gemv_wmma_kernel<<<(waves * 32 + THREADS - 1) / THREADS, THREADS, 0, stream>>>(
            W3, h2, acc3, N, coltiles, kpw, waves);
    }

    pm_sample_kernel<<<D / THREADS, THREADS, 0, stream>>>(
        acc3, ct, qm, noise, sample, absv, lp);

    pm_topk_kernel<<<1, 1024, 0, stream>>>(sample, absv, lp, out, D, kptr);
}
